// EfficientDAGNN_59700045414953
// MI455X (gfx1250) — matrix operand order repaired and checked by reading, not compiled, not run on hardware
//
#include <hip/hip_runtime.h>

typedef __attribute__((ext_vector_type(16))) __bf16 v16bf;
typedef __attribute__((ext_vector_type(8)))  float  v8f;

#define DW    4096
#define BATCH 64

struct Chunk {
  float4 a0, a1, a2, a3;   // 16 activation floats (A fragment, lane-local K subset)
  float4 w0, w1, w2, w3;   // 16 weight floats
  float4 m0, m1, m2, m3;   // 16 mask floats
};

// Loop-invariant uniform bases (SGPR pair) + unsigned 32-bit per-lane element
// offsets -> GVS addressing: global_load_b128 v, vOff, s[base:base+1] offset:imm
__device__ __forceinline__ Chunk load_chunk(const float* __restrict__ aB, unsigned aO,
                                            const float* __restrict__ wB,
                                            const float* __restrict__ mB, unsigned wO) {
  Chunk c;
  // A layout (16x32 bf16): per lane 16 contiguous floats
  c.a0 = *(const float4*)(aB + aO);
  c.a1 = *(const float4*)(aB + aO + 4);
  c.a2 = *(const float4*)(aB + aO + 8);
  c.a3 = *(const float4*)(aB + aO + 12);
  // B layout (32x16 bf16): per lane 16 contiguous floats
  c.w0 = *(const float4*)(wB + wO);
  c.w1 = *(const float4*)(wB + wO + 4);
  c.w2 = *(const float4*)(wB + wO + 8);
  c.w3 = *(const float4*)(wB + wO + 12);
  c.m0 = *(const float4*)(mB + wO);
  c.m1 = *(const float4*)(mB + wO + 4);
  c.m2 = *(const float4*)(mB + wO + 8);
  c.m3 = *(const float4*)(mB + wO + 12);
  return c;
}

__device__ __forceinline__ void pack8(v16bf& d, int o, float4 x, float4 y) {
  d[o + 0] = (__bf16)x.x; d[o + 1] = (__bf16)x.y;
  d[o + 2] = (__bf16)x.z; d[o + 3] = (__bf16)x.w;
  d[o + 4] = (__bf16)y.x; d[o + 5] = (__bf16)y.y;
  d[o + 6] = (__bf16)y.z; d[o + 7] = (__bf16)y.w;
}

__device__ __forceinline__ float4 f4mul(float4 a, float4 b) {
  float4 r; r.x = a.x * b.x; r.y = a.y * b.y; r.z = a.z * b.z; r.w = a.w * b.w;
  return r;
}

__device__ __forceinline__ v8f wmma_step(const Chunk& c, v8f acc) {
  v16bf a, b;
  pack8(a, 0, c.a0, c.a1);
  pack8(a, 8, c.a2, c.a3);
  pack8(b, 0, f4mul(c.w0, c.m0), f4mul(c.w1, c.m1));
  pack8(b, 8, f4mul(c.w2, c.m2), f4mul(c.w3, c.m3));
  return __builtin_amdgcn_wmma_f32_16x16x32_bf16(false, a, false, b,
                                                 (short)0, acc, false, false);
}

// One masked-linear layer: out[64, 4096] = relu(concat(in0,in1)[64, fanin] @ (W*M)^T + b)
// Block: 256 threads = 8 wave32s. Wave w: mi = w&3 -> batch rows [16*mi, 16*mi+16),
// kh = w>>2 -> K half (each half lies entirely in one concat source).
// Block b -> output columns [16b, 16b+16). Two-buffer software pipeline over K,
// fixed uniform bases + u32 lane offsets, K halves reduced through LDS split-barrier.
template <int FANIN>
__global__ __launch_bounds__(256)
void dag_layer(const float* __restrict__ in0, const float* __restrict__ in1,
               const float* __restrict__ W,   const float* __restrict__ Mk,
               const float* __restrict__ bias, float* __restrict__ out)
{
  constexpr unsigned KH = FANIN / 2;           // K elements per wave-group

  const int lane  = threadIdx.x & 31;
  const int wave  = threadIdx.x >> 5;
  const int mi    = wave & 3;                  // batch tile
  const int kh    = wave >> 2;                 // K half
  const int laneN = lane & 15;
  const int hi    = (lane >> 4) & 1;

  const int m   = mi * 16 + laneN;             // batch row (A fragment)
  const int col = blockIdx.x * 16 + laneN;     // output column (B/C/D fragment)

  const unsigned akOff = hi ? 16u : 0u;        // A: lanes<16 K 0..15; lanes>=16 K 16..31
  const unsigned bkOff = hi ? 16u : 0u;        // B: lanes<16 K 0..15; lanes>=16 K 16..31

  const unsigned k0 = (unsigned)kh * KH;

  // Uniform, loop-invariant bases; unsigned 32-bit per-lane element offsets.
  // A K half never straddles the 4096 concat boundary, so the source is fixed.
  const float* aB = (k0 < DW) ? in0 : in1;
  const unsigned aOff = (unsigned)m * DW + ((k0 < DW) ? k0 : (k0 - DW)) + akOff;
  const float* wB = W;
  const float* mB = Mk;
  const unsigned wOff = (unsigned)col * FANIN + k0 + bkOff;  // <= 33.6M elems: u32 bytes OK

  v8f acc = {};

  // ---- two-buffer software-pipelined K loop (64 K / iteration, 2 WMMAs) ----
  Chunk c0 = load_chunk(aB, aOff,       wB, mB, wOff);
  Chunk c1 = load_chunk(aB, aOff + 32u, wB, mB, wOff + 32u);

#pragma unroll 1
  for (unsigned kb = 64u; kb < KH; kb += 64u) {
    acc = wmma_step(c0, acc);
    c0 = load_chunk(aB, aOff + kb, wB, mB, wOff + kb);           // refill in place
    __builtin_prefetch(wB + wOff + kb + 96u, 0, 0);              // stream hint ahead
    __builtin_prefetch(mB + wOff + kb + 96u, 0, 0);
    acc = wmma_step(c1, acc);
    c1 = load_chunk(aB, aOff + kb + 32u, wB, mB, wOff + kb + 32u);
  }
  acc = wmma_step(c0, acc);
  acc = wmma_step(c1, acc);

  // ---- reduce K halves via LDS, then bias + ReLU + store (kh==0 waves) ----
  __shared__ v8f red[4 * 32];                  // 4 KB: one 16x16 f32 tile per mi
  const int ridx = mi * 32 + lane;
  if (kh == 1) red[ridx] = acc;
  __syncthreads();
  if (kh == 0) {
    v8f other = red[ridx];
    const float bv = bias[col];
#pragma unroll
    for (int r = 0; r < 8; ++r) {
      const int orow = mi * 16 + r + hi * 8;
      float y = acc[r] + other[r] + bv;
      out[(size_t)orow * DW + col] = y > 0.0f ? y : 0.0f;
    }
  }
}

extern "C" void kernel_launch(void* const* d_in, const int* in_sizes, int n_in,
                              void* d_out, int out_size, void* d_ws, size_t ws_size,
                              hipStream_t stream) {
  const float* x  = (const float*)d_in[0];
  const float* W1 = (const float*)d_in[1];
  const float* b1 = (const float*)d_in[2];
  const float* M1 = (const float*)d_in[3];
  const float* W2 = (const float*)d_in[4];
  const float* b2 = (const float*)d_in[5];
  const float* M2 = (const float*)d_in[6];
  const float* W3 = (const float*)d_in[7];
  const float* b3 = (const float*)d_in[8];
  const float* M3 = (const float*)d_in[9];
  const float* W4 = (const float*)d_in[10];
  const float* b4 = (const float*)d_in[11];
  const float* M4 = (const float*)d_in[12];

  float* h1 = (float*)d_ws;
  float* h2 = h1 + (size_t)BATCH * DW;
  float* h3 = h2 + (size_t)BATCH * DW;

  dim3 grid(DW / 16), block(256);
  dag_layer<DW><<<grid, block, 0, stream>>>(x,  x,  W1, M1, b1, h1);
  dag_layer<2 * DW><<<grid, block, 0, stream>>>(x,  h1, W2, M2, b2, h2);
  dag_layer<2 * DW><<<grid, block, 0, stream>>>(h1, h2, W3, M3, b3, h3);
  dag_layer<2 * DW><<<grid, block, 0, stream>>>(h2, h3, W4, M4, b4, (float*)d_out);
}
